// ColoringGNN_6210522710529
// MI455X (gfx1250) — compile-verified, matrix operand order checked by reading, not run on hardware
//
#include <hip/hip_runtime.h>
#include <math.h>

// ---------------- problem constants (match reference) ----------------
#define N_NODES  8000
#define N_EDGES  128000
#define N_BATCH  16
#define D_IN     16
#define D_MODEL  256
#define N_HEADS  8
#define C_HEAD   32
#define N_LAYERS 6
#define N_MC     200
#define N_NT     8
#define LRELU_SLOPE 0.2f

typedef __attribute__((ext_vector_type(16))) _Float16 v16h;
typedef __attribute__((ext_vector_type(8)))  float    v8f;

// Monotone order-preserving float<->uint map for atomicMax on floats.
__device__ __forceinline__ unsigned f2o(float f) {
  unsigned b = __float_as_uint(f);
  return (b & 0x80000000u) ? ~b : (b | 0x80000000u);
}
__device__ __forceinline__ float o2f(unsigned o) {
  return (o & 0x80000000u) ? __uint_as_float(o & 0x7fffffffu)
                           : __uint_as_float(~o);
}
#define ORD_NEG_INF 0x007FFFFFu   // f2o(-inf)

// =====================================================================
// WMMA GEMM: Out[M,Nc] = act(A[M,K] * W[K,Nc] + bias[Nc])
// 256 threads = 8 waves arranged 4 (M) x 2 (N); each wave owns a 32x32
// output tile (2x2 WMMA tiles) -> block tile 128x64, K step 32.
// Per K-step per wave: 4 fragments (32 ds_load_b32) -> 4 v_wmma.
// =====================================================================
#define BM 128
#define BN 64
#define BK 32

__global__ __launch_bounds__(256)
void gemm_bias_act_kernel(const float* __restrict__ A, const float* __restrict__ W,
                          const float* __restrict__ bias, float* __restrict__ Out,
                          int M, int K, int Nc, int act) {
  __shared__ _Float16 As[BM][BK + 2];   // row stride 34 halves = 17 dwords (conflict-free)
  __shared__ _Float16 Bs[BN][BK + 2];   // stored transposed: Bs[n][k]

  const int tid    = threadIdx.x;
  const int lane   = tid & 31;
  const int wave   = tid >> 5;
  const int wm     = wave & 3;          // wave row    (32 rows each)
  const int wn     = wave >> 2;         // wave column (32 cols each)
  const int laneHi = (lane >> 4) & 1;
  const int nl     = lane & 15;
  const int m0     = blockIdx.y * BM;
  const int n0     = blockIdx.x * BN;

  // A staging: 128 rows x 32 k = 4096 f32; thread -> (row, 16-float half-row)
  const int ar = tid & 127;
  const int ak = (tid >> 7) << 4;       // 0 or 16
  // B staging: 64 cols x 32 k = 2048 f32; thread -> (col, 8 k-values)
  const int br_ = tid & 63;
  const int bk = (tid >> 6) << 3;       // 0,8,16,24

  v8f acc[2][2] = {};

  for (int kb = 0; kb < K; kb += BK) {
    // ---- stage A tile (f32 -> f16) ----
    {
      const int gm = m0 + ar;
      float tmp[16];
      if (gm < M) {
        const float4* p = reinterpret_cast<const float4*>(A + (size_t)gm * K + kb + ak);
        float4 f0 = p[0], f1 = p[1], f2 = p[2], f3 = p[3];
        tmp[0]=f0.x;  tmp[1]=f0.y;  tmp[2]=f0.z;  tmp[3]=f0.w;
        tmp[4]=f1.x;  tmp[5]=f1.y;  tmp[6]=f1.z;  tmp[7]=f1.w;
        tmp[8]=f2.x;  tmp[9]=f2.y;  tmp[10]=f2.z; tmp[11]=f2.w;
        tmp[12]=f3.x; tmp[13]=f3.y; tmp[14]=f3.z; tmp[15]=f3.w;
        if (kb + BK < K) __builtin_prefetch(A + (size_t)gm * K + kb + BK + ak, 0, 1);
      } else {
        #pragma unroll
        for (int j = 0; j < 16; ++j) tmp[j] = 0.f;
      }
      #pragma unroll
      for (int j = 0; j < 16; ++j) As[ar][ak + j] = (_Float16)tmp[j];
    }
    // ---- stage B tile transposed (f32 -> f16), guard n < Nc ----
    {
      const int gn = n0 + br_;
      #pragma unroll
      for (int j = 0; j < 8; ++j) {
        const int gk = kb + bk + j;
        float v = 0.f;
        if (gn < Nc) v = W[(size_t)gk * Nc + gn];
        Bs[br_][bk + j] = (_Float16)v;
      }
    }
    __syncthreads();

    // ---- build fragments per ISA 16-bit A/B layout (2 halves per dword) ----
    union Frag { v16h v; unsigned u[8]; } fa[2], fb[2];
    #pragma unroll
    for (int p = 0; p < 8; ++p) {
      const int kk = ((p < 4) ? (p << 1) : (16 + ((p - 4) << 1))) + (laneHi << 3);
      fa[0].u[p] = *reinterpret_cast<const unsigned*>(&As[wm * 32 + nl][kk]);
      fa[1].u[p] = *reinterpret_cast<const unsigned*>(&As[wm * 32 + 16 + nl][kk]);
      fb[0].u[p] = *reinterpret_cast<const unsigned*>(&Bs[wn * 32 + nl][kk]);
      fb[1].u[p] = *reinterpret_cast<const unsigned*>(&Bs[wn * 32 + 16 + nl][kk]);
    }
    acc[0][0] = __builtin_amdgcn_wmma_f32_16x16x32_f16(false, fa[0].v, false, fb[0].v,
                                                       (short)0, acc[0][0], false, false);
    acc[0][1] = __builtin_amdgcn_wmma_f32_16x16x32_f16(false, fa[0].v, false, fb[1].v,
                                                       (short)0, acc[0][1], false, false);
    acc[1][0] = __builtin_amdgcn_wmma_f32_16x16x32_f16(false, fa[1].v, false, fb[0].v,
                                                       (short)0, acc[1][0], false, false);
    acc[1][1] = __builtin_amdgcn_wmma_f32_16x16x32_f16(false, fa[1].v, false, fb[1].v,
                                                       (short)0, acc[1][1], false, false);
    __syncthreads();
  }

  // ---- store D (f32 16x16 layout: vgpr r -> M = r + 8*laneHi, lane -> N) ----
  #pragma unroll
  for (int mt = 0; mt < 2; ++mt) {
    #pragma unroll
    for (int nt = 0; nt < 2; ++nt) {
      const int n = n0 + wn * 32 + nt * 16 + nl;
      if (n >= Nc) continue;
      const float bv = bias[n];
      #pragma unroll
      for (int r = 0; r < 8; ++r) {
        const int m = m0 + wm * 32 + mt * 16 + laneHi * 8 + r;
        if (m >= M) continue;
        float v = acc[mt][nt][r] + bv;
        if (act) v = fmaxf(v, 0.f);
        Out[(size_t)m * Nc + n] = v;
      }
    }
  }
}

// =====================================================================
// Node encoder: h = relu(layernorm(x @ enc_W + enc_b)); one block per node.
// =====================================================================
__global__ __launch_bounds__(256)
void encoder_kernel(const float* __restrict__ x, const float* __restrict__ W,
                    const float* __restrict__ b, const float* __restrict__ ls,
                    const float* __restrict__ lb, float* __restrict__ h) {
  __shared__ float xrow[D_IN];
  __shared__ float red[D_MODEL];
  const int n = blockIdx.x, d = threadIdx.x;
  if (d < D_IN) xrow[d] = x[(size_t)n * D_IN + d];
  __syncthreads();
  float acc = b[d];
  #pragma unroll
  for (int k = 0; k < D_IN; ++k) acc += xrow[k] * W[k * D_MODEL + d];
  red[d] = acc; __syncthreads();
  for (int s = 128; s > 0; s >>= 1) { if (d < s) red[d] += red[d + s]; __syncthreads(); }
  const float mu = red[0] * (1.f / D_MODEL);
  __syncthreads();
  const float diff = acc - mu;
  red[d] = diff * diff; __syncthreads();
  for (int s = 128; s > 0; s >>= 1) { if (d < s) red[d] += red[d + s]; __syncthreads(); }
  const float var = red[0] * (1.f / D_MODEL);
  float y = diff * rsqrtf(var + 1e-5f) * ls[d] + lb[d];
  h[(size_t)n * D_MODEL + d] = fmaxf(y, 0.f);
}

// =====================================================================
// Post-GAT: h_new = elu(layernorm(agg + gb)); h = first ? h_new : h + h_new
// =====================================================================
__global__ __launch_bounds__(256)
void ln_elu_res_kernel(const float* __restrict__ agg, const float* __restrict__ gb,
                       const float* __restrict__ ls, const float* __restrict__ lb,
                       float* __restrict__ h, int first) {
  __shared__ float red[D_MODEL];
  const int n = blockIdx.x, d = threadIdx.x;
  const float v = agg[(size_t)n * D_MODEL + d] + gb[d];
  red[d] = v; __syncthreads();
  for (int s = 128; s > 0; s >>= 1) { if (d < s) red[d] += red[d + s]; __syncthreads(); }
  const float mu = red[0] * (1.f / D_MODEL);
  __syncthreads();
  const float diff = v - mu;
  red[d] = diff * diff; __syncthreads();
  for (int s = 128; s > 0; s >>= 1) { if (d < s) red[d] += red[d + s]; __syncthreads(); }
  const float var = red[0] * (1.f / D_MODEL);
  float y = diff * rsqrtf(var + 1e-5f) * ls[d] + lb[d];
  y = (y > 0.f) ? y : (__expf(y) - 1.f);   // ELU(alpha=1)
  const size_t idx = (size_t)n * D_MODEL + d;
  h[idx] = first ? y : (h[idx] + y);
}

// =====================================================================
// Attention passes (edge list = E edges + N self loops), 1 thread per (edge, head)
// =====================================================================
__global__ void attn_logits_kernel(const int* __restrict__ ei,
                                   const float* __restrict__ xl, const float* __restrict__ xr,
                                   const float* __restrict__ att,
                                   float* __restrict__ logits, unsigned* __restrict__ mi,
                                   int Etot) {
  const int t = blockIdx.x * blockDim.x + threadIdx.x;
  if (t >= Etot * N_HEADS) return;
  const int e = t >> 3, hh = t & 7;
  const int src = (e < N_EDGES) ? ei[e] : (e - N_EDGES);
  const int dst = (e < N_EDGES) ? ei[N_EDGES + e] : (e - N_EDGES);
  const float* pl = xl + (size_t)src * D_MODEL + hh * C_HEAD;
  const float* pr = xr + (size_t)dst * D_MODEL + hh * C_HEAD;
  const float* pa = att + hh * C_HEAD;
  float s = 0.f;
  #pragma unroll
  for (int c = 0; c < C_HEAD; ++c) {
    float v = pl[c] + pr[c];
    v = (v > 0.f) ? v : v * LRELU_SLOPE;
    s += v * pa[c];
  }
  logits[t] = s;
  atomicMax(&mi[dst * N_HEADS + hh], f2o(s));
}

__global__ void attn_exp_kernel(const int* __restrict__ ei, float* __restrict__ logits,
                                const unsigned* __restrict__ mi, float* __restrict__ denom,
                                int Etot) {
  const int t = blockIdx.x * blockDim.x + threadIdx.x;
  if (t >= Etot * N_HEADS) return;
  const int e = t >> 3, hh = t & 7;
  const int dst = (e < N_EDGES) ? ei[N_EDGES + e] : (e - N_EDGES);
  const float p = __expf(logits[t] - o2f(mi[dst * N_HEADS + hh]));
  logits[t] = p;
  atomicAdd(&denom[dst * N_HEADS + hh], p);
}

__global__ void attn_agg_kernel(const int* __restrict__ ei, const float* __restrict__ xl,
                                const float* __restrict__ logits, const float* __restrict__ denom,
                                float* __restrict__ agg, int Etot) {
  const int t = blockIdx.x * blockDim.x + threadIdx.x;
  if (t >= Etot * N_HEADS) return;
  const int e = t >> 3, hh = t & 7;
  const int src = (e < N_EDGES) ? ei[e] : (e - N_EDGES);
  const int dst = (e < N_EDGES) ? ei[N_EDGES + e] : (e - N_EDGES);
  const float alpha = logits[t] / denom[dst * N_HEADS + hh];
  const float* pl = xl + (size_t)src * D_MODEL + hh * C_HEAD;
  float* po = agg + (size_t)dst * D_MODEL + hh * C_HEAD;
  #pragma unroll
  for (int c = 0; c < C_HEAD; ++c) atomicAdd(&po[c], pl[c] * alpha);
}

// =====================================================================
// Pooling
// =====================================================================
__global__ void pool_acc_kernel(const float* __restrict__ h, const int* __restrict__ batch,
                                float* __restrict__ sums, unsigned* __restrict__ maxi,
                                float* __restrict__ cnt) {
  const int t = blockIdx.x * blockDim.x + threadIdx.x;
  if (t >= N_NODES * D_MODEL) return;
  const int n = t >> 8, d = t & 255;
  const int b = batch[n];
  const float v = h[t];
  atomicAdd(&sums[b * D_MODEL + d], v);
  atomicMax(&maxi[b * D_MODEL + d], f2o(v));
  if (d == 0) atomicAdd(&cnt[b], 1.0f);
}

__global__ void pool_fin_kernel(const float* __restrict__ sums, const float* __restrict__ cnt,
                                const unsigned* __restrict__ maxi, float* __restrict__ g) {
  const int t = blockIdx.x * blockDim.x + threadIdx.x;
  if (t >= N_BATCH * D_MODEL) return;
  const int b = t >> 8, d = t & 255;
  const float c = fmaxf(cnt[b], 1.f);
  g[b * 2 * D_MODEL + d] = sums[t] / c;
  const float mv = o2f(maxi[t]);
  g[b * 2 * D_MODEL + D_MODEL + d] = (mv > -3.0e38f) ? mv : 0.f;
}

// =====================================================================
// Difficulty head final: sigmoid(zd @ dW2 + db2) * 100
// =====================================================================
__global__ void difficulty_kernel(const float* __restrict__ zd, const float* __restrict__ W,
                                  const float* __restrict__ b, float* __restrict__ out) {
  const int bi = threadIdx.x;
  if (bi >= N_BATCH) return;
  float s = b[0];
  #pragma unroll
  for (int k = 0; k < 64; ++k) s += zd[bi * 64 + k] * W[k];
  out[bi] = 100.f / (1.f + __expf(-s));
}

// ---------------- fills ----------------
__global__ void fill_f32_kernel(float* p, float v, int n) {
  const int i = blockIdx.x * blockDim.x + threadIdx.x;
  if (i < n) p[i] = v;
}
__global__ void fill_u32_kernel(unsigned* p, unsigned v, int n) {
  const int i = blockIdx.x * blockDim.x + threadIdx.x;
  if (i < n) p[i] = v;
}

// =====================================================================
// Host launcher
// =====================================================================
extern "C" void kernel_launch(void* const* d_in, const int* in_sizes, int n_in,
                              void* d_out, int out_size, void* d_ws, size_t ws_size,
                              hipStream_t stream) {
  (void)in_sizes; (void)n_in; (void)out_size; (void)ws_size;
  // ---- inputs (setup_inputs order) ----
  const float* x      = (const float*)d_in[0];
  const int*   ei     = (const int*)  d_in[1];
  const int*   batch  = (const int*)  d_in[2];
  const float* enc_W  = (const float*)d_in[3];
  const float* enc_b  = (const float*)d_in[4];
  const float* enc_ls = (const float*)d_in[5];
  const float* enc_lb = (const float*)d_in[6];
  const float* Wl     = (const float*)d_in[7];
  const float* bl     = (const float*)d_in[8];
  const float* Wr     = (const float*)d_in[9];
  const float* br     = (const float*)d_in[10];
  const float* att    = (const float*)d_in[11];
  const float* gb     = (const float*)d_in[12];
  const float* gls    = (const float*)d_in[13];
  const float* glb    = (const float*)d_in[14];
  const float* cW1 = (const float*)d_in[15]; const float* cb1 = (const float*)d_in[16];
  const float* cW2 = (const float*)d_in[17]; const float* cb2 = (const float*)d_in[18];
  const float* cW3 = (const float*)d_in[19]; const float* cb3 = (const float*)d_in[20];
  const float* chW1 = (const float*)d_in[21]; const float* chb1 = (const float*)d_in[22];
  const float* chW2 = (const float*)d_in[23]; const float* chb2 = (const float*)d_in[24];
  const float* chW3 = (const float*)d_in[25]; const float* chb3 = (const float*)d_in[26];
  const float* tW1 = (const float*)d_in[27]; const float* tb1 = (const float*)d_in[28];
  const float* tW2 = (const float*)d_in[29]; const float* tb2 = (const float*)d_in[30];
  const float* dW1 = (const float*)d_in[31]; const float* db1 = (const float*)d_in[32];
  const float* dW2 = (const float*)d_in[33]; const float* db2 = (const float*)d_in[34];

  // ---- outputs (concatenated flat, return order) ----
  float* out_nc    = (float*)d_out;                     // [N, 200]
  float* out_chrom = out_nc + (size_t)N_NODES * N_MC;   // [B, 200]
  float* out_type  = out_chrom + N_BATCH * N_MC;        // [B, 8]
  float* out_diff  = out_type + N_BATCH * N_NT;         // [B, 1]

  // ---- workspace layout (floats) ----
  float* ws = (float*)d_ws;
  const int ND = N_NODES * D_MODEL;
  const int ET = N_EDGES + N_NODES;
  float*    h      = ws;                 // ND
  float*    xl     = h + ND;             // ND
  float*    xr     = xl + ND;            // ND
  float*    agg    = xr + ND;            // ND
  float*    logits = agg + ND;           // ET*H
  unsigned* mi     = (unsigned*)(logits + (size_t)ET * N_HEADS); // N*H
  float*    denom  = (float*)(mi + N_NODES * N_HEADS);           // N*H
  float*    sums   = denom + N_NODES * N_HEADS;                  // B*D
  float*    cnt    = sums + N_BATCH * D_MODEL;                   // B
  unsigned* maxi   = (unsigned*)(cnt + N_BATCH);                 // B*D
  float*    g      = (float*)(maxi + N_BATCH * D_MODEL);         // B*2D
  float*    zc1    = g + N_BATCH * 2 * D_MODEL;                  // B*256
  float*    zc2    = zc1 + N_BATCH * D_MODEL;                    // B*128
  float*    zt     = zc2 + N_BATCH * 128;                        // B*128
  float*    zd     = zt + N_BATCH * 128;                         // B*64

  auto gemm = [&](const float* A, const float* W, const float* bias, float* Out,
                  int M, int K, int Nc, int act) {
    dim3 grid((Nc + BN - 1) / BN, (M + BM - 1) / BM);
    gemm_bias_act_kernel<<<grid, 256, 0, stream>>>(A, W, bias, Out, M, K, Nc, act);
  };
  auto fillf = [&](float* p, float v, int n) {
    fill_f32_kernel<<<(n + 255) / 256, 256, 0, stream>>>(p, v, n);
  };
  auto fillu = [&](unsigned* p, unsigned v, int n) {
    fill_u32_kernel<<<(n + 255) / 256, 256, 0, stream>>>(p, v, n);
  };

  // 1) node encoder
  encoder_kernel<<<N_NODES, 256, 0, stream>>>(x, enc_W, enc_b, enc_ls, enc_lb, h);

  // 2) GATv2 layers
  const int tot = ET * N_HEADS;
  const int eblocks = (tot + 255) / 256;
  for (int i = 0; i < N_LAYERS; ++i) {
    gemm(h, Wl + (size_t)i * D_MODEL * D_MODEL, bl + i * D_MODEL, xl, N_NODES, D_MODEL, D_MODEL, 0);
    gemm(h, Wr + (size_t)i * D_MODEL * D_MODEL, br + i * D_MODEL, xr, N_NODES, D_MODEL, D_MODEL, 0);
    fillf(agg, 0.f, ND);
    fillu(mi, ORD_NEG_INF, N_NODES * N_HEADS);
    fillf(denom, 0.f, N_NODES * N_HEADS);
    attn_logits_kernel<<<eblocks, 256, 0, stream>>>(ei, xl, xr, att + i * N_HEADS * C_HEAD,
                                                    logits, mi, ET);
    attn_exp_kernel<<<eblocks, 256, 0, stream>>>(ei, logits, mi, denom, ET);
    attn_agg_kernel<<<eblocks, 256, 0, stream>>>(ei, xl, logits, denom, agg, ET);
    ln_elu_res_kernel<<<N_NODES, 256, 0, stream>>>(agg, gb + i * D_MODEL, gls + i * D_MODEL,
                                                   glb + i * D_MODEL, h, i == 0 ? 1 : 0);
  }

  // 3) per-node color head (reuse xl/xr as z1/z2)
  float* z1 = xl; float* z2 = xr;
  gemm(h,  cW1, cb1, z1, N_NODES, D_MODEL, D_MODEL, 1);
  gemm(z1, cW2, cb2, z2, N_NODES, D_MODEL, D_MODEL / 2, 1);
  gemm(z2, cW3, cb3, out_nc, N_NODES, D_MODEL / 2, N_MC, 0);

  // 4) pooling -> g [B, 2D]
  fillf(sums, 0.f, N_BATCH * D_MODEL);
  fillf(cnt, 0.f, N_BATCH);
  fillu(maxi, ORD_NEG_INF, N_BATCH * D_MODEL);
  pool_acc_kernel<<<(ND + 255) / 256, 256, 0, stream>>>(h, batch, sums, maxi, cnt);
  pool_fin_kernel<<<(N_BATCH * D_MODEL + 255) / 256, 256, 0, stream>>>(sums, cnt, maxi, g);

  // 5) chromatic head
  gemm(g,   chW1, chb1, zc1, N_BATCH, 2 * D_MODEL, D_MODEL, 1);
  gemm(zc1, chW2, chb2, zc2, N_BATCH, D_MODEL, D_MODEL / 2, 1);
  gemm(zc2, chW3, chb3, out_chrom, N_BATCH, D_MODEL / 2, N_MC, 0);

  // 6) type head
  gemm(g,  tW1, tb1, zt, N_BATCH, 2 * D_MODEL, 128, 1);
  gemm(zt, tW2, tb2, out_type, N_BATCH, 128, N_NT, 0);

  // 7) difficulty head
  gemm(g, dW1, db1, zd, N_BATCH, 2 * D_MODEL, 64, 1);
  difficulty_kernel<<<1, 32, 0, stream>>>(zd, dW2, db2, out_diff);
}
